// MultiHeadAttention_8881992368333
// MI455X (gfx1250) — matrix operand order repaired and checked by reading, not compiled, not run on hardware
//
#include <hip/hip_runtime.h>

// MI455X (gfx1250) multi-head attention, bf16 WMMA pipeline.
// B=4, T=2048, E=1024, H=16, DK=64.

#define DEVFN __device__ __forceinline__

typedef __attribute__((ext_vector_type(16))) __bf16 bf16x16;
typedef __attribute__((ext_vector_type(8)))  __bf16 bf16x8;
typedef __attribute__((ext_vector_type(4)))  __bf16 bf16x4;
typedef __attribute__((ext_vector_type(8)))  float  f32x8;

constexpr int Bd  = 4;
constexpr int Td  = 2048;
constexpr int Ed  = 1024;
constexpr int Hd  = 16;
constexpr int DKd = 64;
constexpr int Md  = Bd * Td;   // 8192 rows for the projection GEMMs

DEVFN f32x8 wmma_bf16(bf16x16 a, bf16x16 b, f32x8 c) {
  // D = A(16x32 bf16) * B(32x16 bf16) + C(16x16 f32)
  return __builtin_amdgcn_wmma_f32_16x16x32_bf16(
      /*neg_a=*/false, a, /*neg_b=*/false, b,
      /*c_mod=*/(short)0, c, /*reuse_a=*/false, /*reuse_b=*/false);
}

// A fragment (16x32, row-major source, row stride `stride` bf16 elems).
// ISA layout: lane<16 holds row (lane), halves 0..7 = K0..7, 8..15 = K16..23;
// lane>=16 same row, K+8.  => two 16B chunks at kbase+8*hi and kbase+16+8*hi.
DEVFN bf16x16 load_a_frag(const __bf16* base, int stride, int lane, int kbase) {
  const int row = lane & 15;
  const int hi  = lane >> 4;
  const __bf16* p = base + (size_t)row * stride + kbase + 8 * hi;
  bf16x16 r;
  ((bf16x8*)&r)[0] = *(const bf16x8*)(p);
  ((bf16x8*)&r)[1] = *(const bf16x8*)(p + 16);
  return r;
}

// B fragment (32x16) from an N-major buffer (row n contiguous in k, stride
// `stride`).  Lane holds column n = lane&15, K = kbase+8*hi+0..7, kbase+16+8*hi+0..7.
DEVFN bf16x16 load_b_frag(const __bf16* base, int stride, int lane, int kbase) {
  const int n  = lane & 15;
  const int hi = lane >> 4;
  const __bf16* p = base + (size_t)n * stride + kbase + 8 * hi;
  bf16x16 r;
  ((bf16x8*)&r)[0] = *(const bf16x8*)(p);
  ((bf16x8*)&r)[1] = *(const bf16x8*)(p + 16);
  return r;
}

// ---------------------------------------------------------------------------
// Projection GEMM:  Y[m,f] = sum_e X[m,e] * W[f,e] + bias[f]
// Block tile 128(M) x 128(N), K-step 32, 256 threads = 8 waves in a 2x4 grid;
// each wave computes 64x32 via 8 WMMAs per K-step (6 fragment loads).
// XF32:     input X is fp32 (converted to bf16 while staging) else bf16.
// SCATTER:  write bf16 to [B,H,T,DK] layout, else fp32 row-major [M,N].
// ---------------------------------------------------------------------------
template <bool XF32, bool SCATTER>
__global__ __launch_bounds__(256) void proj_gemm_kernel(
    const void* __restrict__ Xv, const float* __restrict__ W,
    const float* __restrict__ bias, void* __restrict__ Out) {
  constexpr int Kdim = Ed;
  constexpr int Ndim = Ed;

  __shared__ __align__(16) __bf16 As[128 * 32];  // [m][k]
  __shared__ __align__(16) __bf16 Bs[128 * 32];  // [n][k] (W is [f][e] row-major)

  const int tid  = threadIdx.x;
  const int lane = tid & 31;
  const int wave = tid >> 5;
  const int m0 = blockIdx.x * 128;
  const int n0 = blockIdx.y * 128;
  const int wm = (wave >> 2) * 64;   // 0 or 64
  const int wn = (wave & 3) * 32;    // 0,32,64,96
  const int hi = lane >> 4;

  const f32x8 zero = {0.f, 0.f, 0.f, 0.f, 0.f, 0.f, 0.f, 0.f};
  f32x8 acc[4][2];
#pragma unroll
  for (int i = 0; i < 4; i++)
#pragma unroll
    for (int j = 0; j < 2; j++) acc[i][j] = zero;

  const float*  Xf = (const float*)Xv;
  const __bf16* Xh = (const __bf16*)Xv;

  for (int k0 = 0; k0 < Kdim; k0 += 32) {
    // Stage X tile (128 rows x 32 cols): 1024 groups of 4 elems, 4/thread.
#pragma unroll
    for (int i = 0; i < 4; i++) {
      const int g = tid + i * 256;
      const int r = g >> 3, c4 = g & 7;
      bf16x4 hv;
      if constexpr (XF32) {
        const float4 f = *(const float4*)(Xf + (size_t)(m0 + r) * Kdim + k0 + c4 * 4);
        hv = bf16x4{(__bf16)f.x, (__bf16)f.y, (__bf16)f.z, (__bf16)f.w};
      } else {
        hv = *(const bf16x4*)(Xh + (size_t)(m0 + r) * Kdim + k0 + c4 * 4);
      }
      *(bf16x4*)(As + r * 32 + c4 * 4) = hv;
    }
    // Stage W tile (128 f-rows x 32 e-cols), fp32 -> bf16.
#pragma unroll
    for (int i = 0; i < 4; i++) {
      const int g = tid + i * 256;
      const int r = g >> 3, c4 = g & 7;
      const float4 f = *(const float4*)(W + (size_t)(n0 + r) * Kdim + k0 + c4 * 4);
      *(bf16x4*)(Bs + r * 32 + c4 * 4) =
          bf16x4{(__bf16)f.x, (__bf16)f.y, (__bf16)f.z, (__bf16)f.w};
    }
    // Prefetch next K-slice into cache while this tile is consumed.
    if (k0 + 32 < Kdim) {
      const int r = tid >> 3, c4 = tid & 7;
      if constexpr (XF32) {
        __builtin_prefetch(Xf + (size_t)(m0 + r) * Kdim + k0 + 32 + c4 * 4, 0, 0);
      } else {
        __builtin_prefetch(Xh + (size_t)(m0 + r) * Kdim + k0 + 32 + c4 * 4, 0, 0);
      }
      __builtin_prefetch(W + (size_t)(n0 + r) * Kdim + k0 + 32 + c4 * 4, 0, 0);
    }
    __syncthreads();

    bf16x16 a[4], b[2];
#pragma unroll
    for (int ms = 0; ms < 4; ms++)
      a[ms] = load_a_frag(As + (wm + ms * 16) * 32, 32, lane, 0);
#pragma unroll
    for (int ns = 0; ns < 2; ns++)
      b[ns] = load_b_frag(Bs + (wn + ns * 16) * 32, 32, lane, 0);
#pragma unroll
    for (int ms = 0; ms < 4; ms++)
#pragma unroll
      for (int ns = 0; ns < 2; ns++)
        acc[ms][ns] = wmma_bf16(a[ms], b[ns], acc[ms][ns]);
    __syncthreads();
  }

  // Epilogue: C layout lane holds col n=lane&15, rows v + 8*hi (v=0..7).
#pragma unroll
  for (int ms = 0; ms < 4; ms++) {
#pragma unroll
    for (int ns = 0; ns < 2; ns++) {
      const int ng = n0 + wn + ns * 16 + (lane & 15);
      const float bv = bias[ng];
#pragma unroll
      for (int v = 0; v < 8; v++) {
        const int mg = m0 + wm + ms * 16 + v + 8 * hi;
        const float val = acc[ms][ns][v] + bv;
        if constexpr (SCATTER) {
          // [B,T,E] row mg, col ng  ->  Q/K/V layout [B,H,T,DK] bf16
          const int bb = mg >> 11;          // / T
          const int tt = mg & (Td - 1);
          const int hh = ng >> 6;           // / DK
          const int dd = ng & (DKd - 1);
          ((__bf16*)Out)[(((size_t)(bb * Hd + hh) * Td + tt) * DKd) + dd] = (__bf16)val;
        } else {
          ((float*)Out)[(size_t)mg * Ndim + ng] = val;
        }
      }
    }
  }
}

// ---------------------------------------------------------------------------
// Flash attention: grid = (T/64, B*H), 128 threads (4 waves, 16 q-rows each).
// 64-key chunks; scores and P*V both via bf16 WMMA; online softmax in fp32.
// Q is pre-scaled by 1/sqrt(DK) = 0.125 (power of two -> exact in bf16).
// ---------------------------------------------------------------------------
__global__ __launch_bounds__(128) void attn_kernel(
    const __bf16* __restrict__ Qp, const __bf16* __restrict__ Kp,
    const __bf16* __restrict__ Vp, const unsigned char* __restrict__ mask,
    __bf16* __restrict__ Xout) {
  __shared__ __align__(16) __bf16 Klds[64 * 64];     // [key][d]
  __shared__ __align__(16) __bf16 Vlds[64 * 64];     // [d][key] (transposed)
  __shared__ __align__(16) __bf16 Plds[4][16 * 64];  // per-wave prob scratch

  const int tid  = threadIdx.x;
  const int lane = tid & 31;
  const int wave = tid >> 5;
  const int hi   = lane >> 4;
  const int bh = blockIdx.y;
  const int b  = bh >> 4;   // / H
  const int h  = bh & 15;
  const int q0 = blockIdx.x * 64 + wave * 16;

  const __bf16* Qbase = Qp + (size_t)bh * Td * DKd;
  const __bf16* Kbase = Kp + (size_t)bh * Td * DKd;
  const __bf16* Vbase = Vp + (size_t)bh * Td * DKd;

  // Q A-fragments for this wave's 16 rows (DK=64 -> two K=32 fragments),
  // pre-scaled by 0.125 (exact: exponent shift only).
  bf16x16 qfrag[2];
  qfrag[0] = load_a_frag(Qbase + (size_t)q0 * DKd, DKd, lane, 0);
  qfrag[1] = load_a_frag(Qbase + (size_t)q0 * DKd, DKd, lane, 32);
#pragma unroll
  for (int kk = 0; kk < 2; kk++)
#pragma unroll
    for (int i = 0; i < 16; i++)
      qfrag[kk][i] = (__bf16)((float)qfrag[kk][i] * 0.125f);

  const f32x8 zero = {0.f, 0.f, 0.f, 0.f, 0.f, 0.f, 0.f, 0.f};
  float mrow[8], lrow[8];
  f32x8 o[4];
#pragma unroll
  for (int v = 0; v < 8; v++) { mrow[v] = -1e30f; lrow[v] = 0.f; }
#pragma unroll
  for (int d = 0; d < 4; d++) o[d] = zero;

  for (int kb = 0; kb < Td; kb += 64) {
    // Stage K tile [key][d] and V tile transposed to [d][key].
    // 64x64 = 4096 elems = 1024 groups of 4, 8 groups per thread.
#pragma unroll
    for (int i = 0; i < 8; i++) {
      const int g = tid + i * 128;
      const int key = g >> 4, c4 = g & 15;
      const bf16x4 kv = *(const bf16x4*)(Kbase + (size_t)(kb + key) * DKd + c4 * 4);
      *(bf16x4*)(Klds + key * 64 + c4 * 4) = kv;
      const bf16x4 vv = *(const bf16x4*)(Vbase + (size_t)(kb + key) * DKd + c4 * 4);
      Vlds[(c4 * 4 + 0) * 64 + key] = vv.x;
      Vlds[(c4 * 4 + 1) * 64 + key] = vv.y;
      Vlds[(c4 * 4 + 2) * 64 + key] = vv.z;
      Vlds[(c4 * 4 + 3) * 64 + key] = vv.w;
    }
    // Prefetch next chunk while computing on this one.
    if (kb + 64 < Td) {
      const int key = tid >> 4, c4 = tid & 15;  // wait: tid up to 127 -> key 0..7
      __builtin_prefetch(Kbase + (size_t)(kb + 64 + key * 8) * DKd + c4 * 4, 0, 0);
      __builtin_prefetch(Vbase + (size_t)(kb + 64 + key * 8) * DKd + c4 * 4, 0, 0);
    }
    __syncthreads();

    // Scores: four 16x16 tiles (64 keys), each = sum of two K=32 WMMAs.
    float pv[4][8];
#pragma unroll
    for (int ns = 0; ns < 4; ns++) {
      f32x8 s = zero;
      s = wmma_bf16(qfrag[0], load_b_frag(Klds + ns * 16 * 64, 64, lane, 0),  s);
      s = wmma_bf16(qfrag[1], load_b_frag(Klds + ns * 16 * 64, 64, lane, 32), s);
      const int keyg = kb + ns * 16 + (lane & 15);
      const bool valid = mask[(size_t)b * Td + keyg] != 0;
#pragma unroll
      for (int v = 0; v < 8; v++) pv[ns][v] = valid ? s[v] : -1e30f;
    }

    // Online softmax.  Lanes sharing hi hold the same 8 rows; the 16 lanes
    // with the same hi cover the 16 columns of each tile.
    float alpha[8];
#pragma unroll
    for (int v = 0; v < 8; v++) {
      float r = fmaxf(fmaxf(pv[0][v], pv[1][v]), fmaxf(pv[2][v], pv[3][v]));
      r = fmaxf(r, __shfl_xor(r, 1, 32));
      r = fmaxf(r, __shfl_xor(r, 2, 32));
      r = fmaxf(r, __shfl_xor(r, 4, 32));
      r = fmaxf(r, __shfl_xor(r, 8, 32));
      const float mn = fmaxf(mrow[v], r);
      alpha[v] = __expf(mrow[v] - mn);
      mrow[v] = mn;
    }
#pragma unroll
    for (int v = 0; v < 8; v++) {
      float rs = 0.f;
#pragma unroll
      for (int ns = 0; ns < 4; ns++) {
        const float p = __expf(pv[ns][v] - mrow[v]);
        pv[ns][v] = p;
        rs += p;
      }
      rs += __shfl_xor(rs, 1, 32);
      rs += __shfl_xor(rs, 2, 32);
      rs += __shfl_xor(rs, 4, 32);
      rs += __shfl_xor(rs, 8, 32);
      lrow[v] = lrow[v] * alpha[v] + rs;
    }
#pragma unroll
    for (int d = 0; d < 4; d++)
#pragma unroll
      for (int v = 0; v < 8; v++) o[d][v] *= alpha[v];

    // C-layout -> A-layout via per-wave LDS scratch (16x64 probs).
    __bf16* P = &Plds[wave][0];
#pragma unroll
    for (int ns = 0; ns < 4; ns++)
#pragma unroll
      for (int v = 0; v < 8; v++)
        P[(v + 8 * hi) * 64 + ns * 16 + (lane & 15)] = (__bf16)pv[ns][v];
    __syncthreads();

    // O += P(16x64) * V(64x64): two A-frags, four d-subtiles.
    bf16x16 pfrag[2];
    pfrag[0] = load_a_frag(P, 64, lane, 0);
    pfrag[1] = load_a_frag(P, 64, lane, 32);
#pragma unroll
    for (int d = 0; d < 4; d++) {
      o[d] = wmma_bf16(pfrag[0], load_b_frag(Vlds + d * 16 * 64, 64, lane, 0),  o[d]);
      o[d] = wmma_bf16(pfrag[1], load_b_frag(Vlds + d * 16 * 64, 64, lane, 32), o[d]);
    }
    __syncthreads();  // all waves done with Klds/Vlds before restaging
  }

  // Normalize and write x in [B,T,E] bf16 (E col = h*DK + d*16 + lane&15).
#pragma unroll
  for (int v = 0; v < 8; v++) {
    const float inv = lrow[v] > 0.f ? 1.f / lrow[v] : 0.f;
    const int t = q0 + v + 8 * hi;
#pragma unroll
    for (int d = 0; d < 4; d++) {
      const int e = h * DKd + d * 16 + (lane & 15);
      Xout[((size_t)b * Td + t) * Ed + e] = (__bf16)(o[d][v] * inv);
    }
  }
}

// ---------------------------------------------------------------------------
extern "C" void kernel_launch(void* const* d_in, const int* in_sizes, int n_in,
                              void* d_out, int out_size, void* d_ws, size_t ws_size,
                              hipStream_t stream) {
  (void)in_sizes; (void)n_in; (void)out_size; (void)ws_size;

  const float* query = (const float*)d_in[0];
  const float* key_  = (const float*)d_in[1];
  const float* value = (const float*)d_in[2];
  const unsigned char* mask = (const unsigned char*)d_in[3];
  const float* w_q = (const float*)d_in[4];
  const float* b_q = (const float*)d_in[5];
  const float* w_k = (const float*)d_in[6];
  const float* b_k = (const float*)d_in[7];
  const float* w_v = (const float*)d_in[8];
  const float* b_v = (const float*)d_in[9];
  const float* w_o = (const float*)d_in[10];
  const float* b_o = (const float*)d_in[11];

  // Workspace: Q,K,V in [B,H,T,DK] bf16 (16 MB each) + attention out (16 MB).
  const size_t nElem = (size_t)Bd * Hd * Td * DKd;  // 8,388,608
  __bf16* Qp    = (__bf16*)d_ws;
  __bf16* Kp    = Qp + nElem;
  __bf16* Vp    = Kp + nElem;
  __bf16* Xattn = Vp + nElem;

  const dim3 gemm_grid(Md / 128, Ed / 128);  // (64, 8)
  const dim3 gemm_block(256);

  proj_gemm_kernel<true, true><<<gemm_grid, gemm_block, 0, stream>>>(query, w_q, b_q, Qp);
  proj_gemm_kernel<true, true><<<gemm_grid, gemm_block, 0, stream>>>(key_,  w_k, b_k, Kp);
  proj_gemm_kernel<true, true><<<gemm_grid, gemm_block, 0, stream>>>(value, w_v, b_v, Vp);

  const dim3 attn_grid(Td / 64, Bd * Hd);  // (32, 64)
  attn_kernel<<<attn_grid, dim3(128), 0, stream>>>(Qp, Kp, Vp, mask, Xattn);

  proj_gemm_kernel<false, false><<<gemm_grid, gemm_block, 0, stream>>>(Xattn, w_o, b_o, d_out);
}
